// CausalSelfAttention_29549374996598
// MI455X (gfx1250) — matrix operand order repaired and checked by reading, not compiled, not run on hardware
//
#include <hip/hip_runtime.h>
#include <hip/hip_bf16.h>

// ---------------------------------------------------------------------------
// Causal self-attention for MI455X (gfx1250), bf16 WMMA (f32 accumulate).
//
// Roofline: ~60 GFLOP vs ~100MB HBM traffic -> matrix-unit bound. FP32 WMMA
// is only 16x16x4; bf16 WMMA is 16x16x32 (8x MACs/inst) -> convert fp32
// operands to bf16 in LDS staging, accumulate f32.
// V is pre-transposed once into workspace so the attention inner loop stages
// both K and V^T tiles with b128 LDS stores (no scalar transpose in the loop).
// ---------------------------------------------------------------------------

typedef float  v8f   __attribute__((ext_vector_type(8)));
typedef __bf16 v16bf __attribute__((ext_vector_type(16)));
typedef __bf16 v8bf  __attribute__((ext_vector_type(8)));
typedef __bf16 v4bf  __attribute__((ext_vector_type(4)));

#define CAT16(lo, hi) __builtin_shufflevector((lo), (hi), 0,1,2,3,4,5,6,7,8,9,10,11,12,13,14,15)
#define WMMA_BF16(a, b, c) \
  __builtin_amdgcn_wmma_f32_16x16x32_bf16(false, (a), false, (b), (short)0, (c), false, false)

// ====================== GEMM: out[M,N] = A[M,1024] * W[N,1024]^T + bias ====
// Block tile 128x128, 8 waves (4x2), wave tile 32x64 (2x4 WMMA accumulators).
// QKV=true : epilogue scales Q columns (<1024) by 1/sqrt(64), writes bf16.
// QKV=false: epilogue writes f32 (final output).

#define GLDK 40  // 32 K + 8 pad (bf16 elems); 80B row stride, 16B aligned

template <bool QKV>
__global__ __launch_bounds__(256) void gemm_bf16_wmma(
    const float* __restrict__ A, const float* __restrict__ W,
    const float* __restrict__ bias, __bf16* __restrict__ outBf,
    float* __restrict__ outF, int Ntot) {
  constexpr int K = 1024;
  __shared__ __bf16 ldsA[128 * GLDK];
  __shared__ __bf16 ldsB[128 * GLDK];

  const int tid  = threadIdx.x;
  const int lane = tid & 31;
  const int w    = tid >> 5;
  const int wm   = w >> 1;      // 0..3
  const int wn   = w & 1;       // 0..1
  const int ln15 = lane & 15;
  const int lh   = lane >> 4;
  const int m0   = blockIdx.y * 128;
  const int n0   = blockIdx.x * 128;

  v8f vzero = {};
  v8f acc[2][4];
#pragma unroll
  for (int ms = 0; ms < 2; ms++)
#pragma unroll
    for (int ns = 0; ns < 4; ns++) acc[ms][ns] = vzero;

  for (int k0 = 0; k0 < K; k0 += 32) {
    // ---- stage A and B tiles (fp32 -> bf16), 1024 4-elem chunks each ----
#pragma unroll
    for (int i = 0; i < 4; i++) {
      int cid = tid * 4 + i;
      int row = cid >> 3;
      int kc  = (cid & 7) * 4;
      const float* ap = A + (size_t)(m0 + row) * K + k0 + kc;
      const float* bp = W + (size_t)(n0 + row) * K + k0 + kc;
      if (k0 + 32 < K) {  // prefetch next K-step (global_prefetch_b8)
        __builtin_prefetch(ap + 32, 0, 0);
        __builtin_prefetch(bp + 32, 0, 0);
      }
      float4 av = *(const float4*)ap;
      v4bf ab = {(__bf16)av.x, (__bf16)av.y, (__bf16)av.z, (__bf16)av.w};
      *(v4bf*)&ldsA[row * GLDK + kc] = ab;
      float4 bv = *(const float4*)bp;
      v4bf bb = {(__bf16)bv.x, (__bf16)bv.y, (__bf16)bv.z, (__bf16)bv.w};
      *(v4bf*)&ldsB[row * GLDK + kc] = bb;
    }
    __syncthreads();

    // ---- fragments + 8 WMMAs ----
    const int koff = lh * 8;
    v16bf afrag[2], bfrag[4];
#pragma unroll
    for (int ms = 0; ms < 2; ms++) {
      int row = wm * 32 + ms * 16 + ln15;
      v8bf lo = *(const v8bf*)&ldsA[row * GLDK + koff];
      v8bf hi = *(const v8bf*)&ldsA[row * GLDK + koff + 16];
      afrag[ms] = CAT16(lo, hi);
    }
#pragma unroll
    for (int ns = 0; ns < 4; ns++) {
      int row = wn * 64 + ns * 16 + ln15;
      int off = lh * 8;
      v8bf lo = *(const v8bf*)&ldsB[row * GLDK + off];
      v8bf hi = *(const v8bf*)&ldsB[row * GLDK + off + 16];
      bfrag[ns] = CAT16(lo, hi);
    }
#pragma unroll
    for (int ms = 0; ms < 2; ms++)
#pragma unroll
      for (int ns = 0; ns < 4; ns++)
        acc[ms][ns] = WMMA_BF16(afrag[ms], bfrag[ns], acc[ms][ns]);
    __syncthreads();
  }

  // ---- epilogue: C layout lane holds M = r + 8*(lane>>4), N = lane&15 ----
#pragma unroll
  for (int ms = 0; ms < 2; ms++)
#pragma unroll
    for (int ns = 0; ns < 4; ns++)
#pragma unroll
      for (int r = 0; r < 8; r++) {
        int M   = r + lh * 8;
        int row = m0 + wm * 32 + ms * 16 + M;
        int col = n0 + wn * 64 + ns * 16 + ln15;
        float v = acc[ms][ns][r] + bias[col];
        if (QKV) {
          if (col < 1024) v *= 0.125f;  // fold 1/sqrt(HEAD_DIM) into Q
          outBf[(size_t)row * Ntot + col] = (__bf16)v;
        } else {
          outF[(size_t)row * Ntot + col] = v;
        }
      }
}

// ====================== V transpose: qkv V region -> vT[B*H*DH, S] =========
// One-shot coalesced transpose through LDS so the attention key loop never
// does scalar transposes. Block: 64 keys x one head.

__global__ __launch_bounds__(128) void vtrans_kernel(
    const __bf16* __restrict__ qkv, __bf16* __restrict__ vT) {
  constexpr int S = 2048, E = 3072, DH = 64;
  const int tid = threadIdx.x;
  const int k0  = blockIdx.x * 64;
  const int h   = blockIdx.y;
  const int b   = blockIdx.z;

  __shared__ __bf16 lds[64 * 72];  // [key][d]

#pragma unroll
  for (int i = 0; i < 4; i++) {
    int cid = tid * 4 + i;  // 512 chunks = 64 keys x 8 d-chunks
    int row = cid >> 3;
    int dc  = (cid & 7) * 8;
    *(v8bf*)&lds[row * 72 + dc] =
        *(const v8bf*)(qkv + (size_t)(b * S + k0 + row) * E + 2048 + h * DH + dc);
  }
  __syncthreads();
#pragma unroll
  for (int i = 0; i < 4; i++) {
    int cid = tid * 4 + i;  // 512 chunks = 64 d-rows x 8 key-chunks
    int d   = cid >> 3;
    int kc  = (cid & 7) * 8;
    v8bf o;
#pragma unroll
    for (int j = 0; j < 8; j++) o[j] = lds[(kc + j) * 72 + d];
    *(v8bf*)(vT + (size_t)((b * 16 + h) * DH + d) * S + k0 + kc) = o;
  }
}

// ====================== Flash attention per (b, h, 64-query block) =========
// 4 waves x 16 query rows; key blocks of 64; online softmax via LDS.

__global__ __launch_bounds__(128) void attn_kernel(
    const __bf16* __restrict__ qkv, const __bf16* __restrict__ vT,
    float* __restrict__ attn) {
  constexpr int S = 2048, D = 1024, E = 3072, DH = 64;
  const int tid  = threadIdx.x;
  const int lane = tid & 31;
  const int wv   = tid >> 5;     // wave 0..3
  const int ln15 = lane & 15;
  const int lh   = lane >> 4;
  const int koff = lh * 8;
  const int qb   = blockIdx.x;   // 0..31
  const int h    = blockIdx.y;
  const int b    = blockIdx.z;
  const int q0   = qb * 64;

  __shared__ __bf16 ldsK[64 * 72];    // [key][d]  (B-frag for Q*K^T)
  __shared__ __bf16 ldsVT[64 * 72];   // [d][key]  (B-frag for P*V)
  __shared__ float  ldsS[4][16 * 68]; // wave-private score tiles
  __shared__ __bf16 ldsP[4][16 * 72]; // wave-private probabilities (A-frag)
  __shared__ float  ldsAlpha[4][16];
  __shared__ float  ldsLsum[4][16];

  // ---- Q fragments, 2 K-steps over Dh=64 (scale already folded in) ----
  v16bf qf[2];
  {
    const size_t qrow = (size_t)(b * S + q0 + wv * 16 + ln15) * E + h * DH;
#pragma unroll
    for (int ks = 0; ks < 2; ks++) {
      v8bf lo = *(const v8bf*)(qkv + qrow + ks * 32 + koff);
      v8bf hi = *(const v8bf*)(qkv + qrow + ks * 32 + koff + 16);
      qf[ks] = CAT16(lo, hi);
    }
  }

  v8f vzero = {};
  v8f acco[4];
#pragma unroll
  for (int t = 0; t < 4; t++) acco[t] = vzero;
  float m_run = -3.0e38f, l_run = 0.0f;

  const size_t vtHead = (size_t)(b * 16 + h) * DH * S;

  const int nkb = qb + 1;  // causal: only key blocks with k0 <= q_last
  for (int kb = 0; kb < nkb; kb++) {
    const int k0 = kb * 64;
    // ---- stage K [key][d] and V^T [d][key]; all b128 LDS stores ----
#pragma unroll
    for (int i = 0; i < 4; i++) {
      int cid = tid * 4 + i;           // 512 chunks = 64 rows x 8
      int row = cid >> 3;
      int dc  = (cid & 7) * 8;
      *(v8bf*)&ldsK[row * 72 + dc] =
          *(const v8bf*)(qkv + (size_t)(b * S + k0 + row) * E + D + h * DH + dc);
      // here reuse mapping: row = d-row of V^T, dc = key-chunk
      *(v8bf*)&ldsVT[row * 72 + dc] =
          *(const v8bf*)(vT + vtHead + (size_t)row * S + k0 + dc);
    }
    __syncthreads();

    // ---- scores: 4 tiles of 16x16, masked, -> wave-private LDS ----
    float* Sw = &ldsS[wv][0];
#pragma unroll
    for (int nt = 0; nt < 4; nt++) {
      v8f sc = vzero;
#pragma unroll
      for (int ks = 0; ks < 2; ks++) {
        int krow = nt * 16 + ln15;
        int doff = ks * 32 + lh * 8;
        v8bf lo = *(const v8bf*)&ldsK[krow * 72 + doff];
        v8bf hi = *(const v8bf*)&ldsK[krow * 72 + doff + 16];
        v16bf bf = CAT16(lo, hi);
        sc = WMMA_BF16(qf[ks], bf, sc);
      }
#pragma unroll
      for (int r = 0; r < 8; r++) {
        int M = r + lh * 8;
        int q = q0 + wv * 16 + M;
        int k = k0 + nt * 16 + ln15;
        Sw[M * 68 + nt * 16 + ln15] = (k > q) ? -3.0e38f : sc[r];
      }
    }

    // ---- online softmax: lanes 0-15, one query row each ----
    if (lane < 16) {
      float rmax = -3.0e38f;
#pragma unroll
      for (int j = 0; j < 64; j++) rmax = fmaxf(rmax, Sw[lane * 68 + j]);
      float mnew  = fmaxf(m_run, rmax);
      float alpha = __expf(m_run - mnew);
      float psum  = 0.0f;
#pragma unroll
      for (int j = 0; j < 64; j++) {
        float p = __expf(Sw[lane * 68 + j] - mnew);
        psum += p;
        ldsP[wv][lane * 72 + j] = (__bf16)p;
      }
      l_run = l_run * alpha + psum;
      m_run = mnew;
      ldsAlpha[wv][lane] = alpha;
    }

    // ---- rescale running output accumulators ----
#pragma unroll
    for (int r = 0; r < 8; r++) {
      float a = ldsAlpha[wv][r + lh * 8];
#pragma unroll
      for (int t = 0; t < 4; t++) acco[t][r] *= a;
    }

    // ---- P*V: A = P (16x64), B = V^T tiles ----
#pragma unroll
    for (int ks = 0; ks < 2; ks++) {
      v8bf plo = *(const v8bf*)&ldsP[wv][ln15 * 72 + ks * 32 + koff];
      v8bf phi = *(const v8bf*)&ldsP[wv][ln15 * 72 + ks * 32 + koff + 16];
      v16bf pa = CAT16(plo, phi);
#pragma unroll
      for (int t = 0; t < 4; t++) {
        int d   = t * 16 + ln15;
        int key = ks * 32 + lh * 8;
        v8bf blo = *(const v8bf*)&ldsVT[d * 72 + key];
        v8bf bhi = *(const v8bf*)&ldsVT[d * 72 + key + 16];
        v16bf vb = CAT16(blo, bhi);
        acco[t] = WMMA_BF16(pa, vb, acco[t]);
      }
    }
    __syncthreads();
  }

  // ---- normalize and write attn (f32, head-interleaved [B,S,D]) ----
  if (lane < 16) ldsLsum[wv][lane] = l_run;
#pragma unroll
  for (int t = 0; t < 4; t++)
#pragma unroll
    for (int r = 0; r < 8; r++) {
      int M = r + lh * 8;
      float inv = 1.0f / ldsLsum[wv][M];
      size_t row = (size_t)(b * S + q0 + wv * 16 + M);
      attn[row * D + h * DH + t * 16 + ln15] = acco[t][r] * inv;
    }
}

// ===========================================================================
extern "C" void kernel_launch(void* const* d_in, const int* in_sizes, int n_in,
                              void* d_out, int out_size, void* d_ws, size_t ws_size,
                              hipStream_t stream) {
  const float* x     = (const float*)d_in[0];  // [2,2048,1024]
  const float* w_qkv = (const float*)d_in[1];  // [3072,1024]
  const float* b_qkv = (const float*)d_in[2];  // [3072]
  const float* w_out = (const float*)d_in[3];  // [1024,1024]
  const float* b_out = (const float*)d_in[4];  // [1024]
  float* out = (float*)d_out;                  // [2,2048,1024]

  // workspace layout (bytes):
  //   qkv  bf16 [4096,3072] : 25,165,824
  //   attn f32  [4096,1024] : 16,777,216
  //   vT   bf16 [2*16*64,2048] : 8,388,608     total ~50.4 MB
  __bf16* qkv = (__bf16*)d_ws;
  float* attn = (float*)((char*)d_ws + (size_t)4096 * 3072 * 2);
  __bf16* vT  = (__bf16*)((char*)d_ws + (size_t)4096 * 3072 * 2 +
                          (size_t)4096 * 1024 * 4);

  // 1) QKV projection (writes bf16, scales Q by 1/8)
  gemm_bf16_wmma<true><<<dim3(3072 / 128, 4096 / 128), dim3(256), 0, stream>>>(
      x, w_qkv, b_qkv, qkv, nullptr, 3072);

  // 2) one-shot V transpose -> vT[B*H*DH, S]
  vtrans_kernel<<<dim3(2048 / 64, 16, 2), dim3(128), 0, stream>>>(qkv, vT);

  // 3) causal flash attention per (64-q-block, head, batch)
  attn_kernel<<<dim3(2048 / 64, 16, 2), dim3(128), 0, stream>>>(qkv, vT, attn);

  // 4) output projection (writes f32)
  gemm_bf16_wmma<false><<<dim3(1024 / 128, 4096 / 128), dim3(256), 0, stream>>>(
      attn, w_out, b_out, nullptr, out, 1024);
}
